// BertLayer_2989297238363
// MI455X (gfx1250) — matrix operand order repaired and checked by reading, not compiled, not run on hardware
//
#include <hip/hip_runtime.h>
#include <math.h>

typedef __bf16 bf16;
typedef __attribute__((ext_vector_type(4)))  bf16  v4bf;
typedef __attribute__((ext_vector_type(8)))  bf16  v8bf;
typedef __attribute__((ext_vector_type(16))) bf16  v16bf;
typedef __attribute__((ext_vector_type(8)))  float v8f;
typedef __attribute__((ext_vector_type(4)))  unsigned u32x4;
typedef __attribute__((ext_vector_type(4)))  int   i32x4;
typedef __attribute__((ext_vector_type(8)))  int   i32x8;

#if defined(__AMDGCN__) && __has_builtin(__builtin_amdgcn_tensor_load_to_lds) && __has_builtin(__builtin_amdgcn_s_wait_tensorcnt)
#define USE_TDM 1
#else
#define USE_TDM 0
#endif

#define DEV static __device__ __forceinline__

DEV v16bf cat16(v8bf a, v8bf b){
  return __builtin_shufflevector(a, b, 0,1,2,3,4,5,6,7,8,9,10,11,12,13,14,15);
}

// A-matrix fragment 16x32 (MxK), row-major source, leading dim ld (elements).
// ISA: lanes 0-15 hold K=0..7 / K=16..23 ; lanes 16-31 hold K=8..15 / K=24..31.
DEV v16bf frag_a(const bf16* p, int ld, int lane){
  int r  = lane & 15;
  int kh = (lane >> 4) << 3;
  const bf16* q = p + r * ld + kh;
  return cat16(*(const v8bf*)q, *(const v8bf*)(q + 16));
}

// B-matrix fragment 32x16 (KxN), source rows are the N dimension ([N][K], K contig).
// ISA: lanes 0-15 hold K=0..7 / K=16..23 ; lanes 16-31 hold K=8..15 / K=24..31.
DEV v16bf frag_b(const bf16* p, int ld, int lane){
  int r  = lane & 15;
  int kh = (lane >> 4) << 3;
  const bf16* q = p + r * ld + kh;
  return cat16(*(const v8bf*)q, *(const v8bf*)(q + 16));
}

// B-matrix fragment 32x16 where source is row-major [K][N] (strided columns).
DEV v16bf frag_b_col(const bf16* p, int ld, int lane){
  int n  = lane & 15;
  int k0 = (lane >> 4) << 3;
  v16bf f;
#pragma unroll
  for (int i = 0; i < 16; i++) f[i] = p[(k0 + i + (i >= 8 ? 8 : 0)) * ld + n];
  return f;
}

DEV v8f wmma_bf16(v16bf a, v16bf b, v8f c){
  return __builtin_amdgcn_wmma_f32_16x16x32_bf16(false, a, false, b, (short)0, c, false, false);
}

#if USE_TDM
// Issue one TDM 2-D tile load (bf16 elements). D# per CDNA5 ISA ch.8:
// group0: count=1 | lds_addr | global_addr(57) | type=2
// group1: data_size=2B | tensor_dim0/1 | tile_dim0/1 | tensor_dim0_stride
DEV void tdm_load_2d(unsigned lds_byte, const void* gptr,
                     unsigned dim0, unsigned dim1,
                     unsigned tile0, unsigned tile1, unsigned stride0)
{
  unsigned long long ga = (unsigned long long)(size_t)gptr;
  u32x4 g0;
  g0[0] = 1u;                                           // count=1, user mode
  g0[1] = lds_byte;                                     // lds_addr
  g0[2] = (unsigned)ga;                                 // global_addr[31:0]
  g0[3] = ((unsigned)(ga >> 32) & 0x01FFFFFFu) | (2u << 30);  // addr[56:32] | type=2
  i32x8 g1;
  g1[0] = (int)(1u << 16);                              // data_size=1 -> 2 bytes
  g1[1] = (int)((dim0 & 0xFFFFu) << 16);                // tensor_dim0[15:0]
  g1[2] = (int)((dim0 >> 16) | ((dim1 & 0xFFFFu) << 16));   // dim0 hi | dim1 lo
  g1[3] = (int)((dim1 >> 16) | (tile0 << 16));          // dim1 hi | tile_dim0
  g1[4] = (int)tile1;                                   // tile_dim1 (tile_dim2=0)
  g1[5] = (int)stride0;                                 // tensor_dim0_stride[31:0]
  g1[6] = 0;
  g1[7] = 0;
  i32x4 z4 = {0, 0, 0, 0};
#if __clang_major__ >= 23
  i32x8 z8 = {0, 0, 0, 0, 0, 0, 0, 0};
  __builtin_amdgcn_tensor_load_to_lds(g0, g1, z4, z4, z8, 0);
#else
  __builtin_amdgcn_tensor_load_to_lds(g0, g1, z4, z4, 0);
#endif
}
DEV unsigned lds_off(const void* p){ return (unsigned)(size_t)p; }  // LDS flat addr low 32 = offset
#endif

// ---------------------------------------------------------------- convert ---
__global__ __launch_bounds__(256) void cvt_kernel(const float* __restrict__ in,
                                                  bf16* __restrict__ out, int n4){
  int i = blockIdx.x * 256 + threadIdx.x;
  if (i < n4){
    float4 v = ((const float4*)in)[i];
    v4bf o = {(bf16)v.x, (bf16)v.y, (bf16)v.z, (bf16)v.w};
    ((v4bf*)out)[i] = o;
  }
}

// ------------------------------------------------------------------- GEMM ---
// C[M,N] = A[M,K] * W[N,K]^T + bias ; MODE 0: bf16 out, 1: f32 out, 2: gelu+bf16
template<int MODE>
__global__ __launch_bounds__(256) void gemm_kernel(
    const bf16* __restrict__ A, const bf16* __restrict__ W,
    const float* __restrict__ bias, void* __restrict__ out,
    int M, int N, int K)
{
#if USE_TDM
  __shared__ __align__(16) bf16 As[2][128 * 32];
  __shared__ __align__(16) bf16 Ws[2][128 * 32];
#else
  __shared__ __align__(16) bf16 As[1][128 * 32];
  __shared__ __align__(16) bf16 Ws[1][128 * 32];
#endif
  const int tiles_n = N >> 7;
  const int bm = blockIdx.x / tiles_n;
  const int bn = blockIdx.x % tiles_n;
  const int tid = threadIdx.x, lane = tid & 31, wave = tid >> 5;
  const int wm = wave >> 1;           // 0..3 -> 32 rows
  const int wn = wave & 1;            // 0..1 -> 64 cols

  v8f acc[2][4];
#pragma unroll
  for (int i = 0; i < 2; i++)
#pragma unroll
    for (int j = 0; j < 4; j++) acc[i][j] = {};

  const bf16* Abase = A + (size_t)(bm * 128) * K;
  const bf16* Wbase = W + (size_t)(bn * 128) * K;
  int buf = 0;

#if USE_TDM
  if (wave == 0){
    tdm_load_2d(lds_off(As[0]), Abase, K, M, 32, 128, K);
    tdm_load_2d(lds_off(Ws[0]), Wbase, K, N, 32, 128, K);
    __builtin_amdgcn_s_wait_tensorcnt(0);
  }
  __syncthreads();
#else
  const int lr = tid >> 1;
  const int lc = (tid & 1) << 4;
#endif

  for (int k0 = 0; k0 < K; k0 += 32){
#if USE_TDM
    if (wave == 0 && k0 + 32 < K){      // prefetch next tile via TDM (double buffer)
      tdm_load_2d(lds_off(As[buf ^ 1]), Abase + k0 + 32, K, M, 32, 128, K);
      tdm_load_2d(lds_off(Ws[buf ^ 1]), Wbase + k0 + 32, K, N, 32, 128, K);
    }
#else
    const v8bf* ga = (const v8bf*)(Abase + (size_t)lr * K + k0 + lc);
    *(v8bf*)(As[0] + lr * 32 + lc)     = ga[0];
    *(v8bf*)(As[0] + lr * 32 + lc + 8) = ga[1];
    const v8bf* gw = (const v8bf*)(Wbase + (size_t)lr * K + k0 + lc);
    *(v8bf*)(Ws[0] + lr * 32 + lc)     = gw[0];
    *(v8bf*)(Ws[0] + lr * 32 + lc + 8) = gw[1];
    __syncthreads();
#endif

    v16bf af[2], bfg[4];
#pragma unroll
    for (int i = 0; i < 2; i++) af[i] = frag_a(As[buf] + (wm * 32 + i * 16) * 32, 32, lane);
#pragma unroll
    for (int j = 0; j < 4; j++) bfg[j] = frag_b(Ws[buf] + (wn * 64 + j * 16) * 32, 32, lane);
#pragma unroll
    for (int i = 0; i < 2; i++)
#pragma unroll
      for (int j = 0; j < 4; j++) acc[i][j] = wmma_bf16(af[i], bfg[j], acc[i][j]);

#if USE_TDM
    if (wave == 0) __builtin_amdgcn_s_wait_tensorcnt(0);
    __syncthreads();
    buf ^= 1;
#else
    __syncthreads();
#endif
  }

  const int hl = lane >> 4, c16 = lane & 15;
#pragma unroll
  for (int j = 0; j < 4; j++){
    int gcol = bn * 128 + wn * 64 + j * 16 + c16;
    float bv = bias[gcol];
#pragma unroll
    for (int i = 0; i < 2; i++){
#pragma unroll
      for (int e = 0; e < 8; e++){
        int grow = bm * 128 + wm * 32 + i * 16 + e + 8 * hl;
        float v = acc[i][j][e] + bv;
        if (MODE == 2) v = 0.5f * v * (1.0f + erff(v * 0.70710678118f));
        if (MODE == 1) ((float*)out)[(size_t)grow * N + gcol] = v;
        else           ((bf16*)out)[(size_t)grow * N + gcol] = (bf16)v;
      }
    }
  }
}

// -------------------------------------------------------- flash attention ---
// q = k = v = qkv (reference reuses the q projection). One block = 128 query
// rows of one (b,h); 8 waves x 16 rows. Streams 32-key blocks through LDS.
__global__ __launch_bounds__(256) void attention_kernel(
    const bf16* __restrict__ qkv, const float* __restrict__ mask,
    bf16* __restrict__ attn)
{
  constexpr int S = 2048, D = 1024, H = 16, DH = 64;
  const int nqb = S / 128;
  const int bh  = blockIdx.x / nqb;
  const int qb0 = (blockIdx.x % nqb) * 128;
  const int b = bh / H, h = bh % H;
  const bf16* base = qkv + (size_t)b * S * D + h * DH;   // row stride D
  const float* mb = mask + (size_t)b * S;
  const int tid = threadIdx.x, lane = tid & 31, wave = tid >> 5;
  const int hl = lane >> 4, c16 = lane & 15;
  const int q0 = qb0 + wave * 16;

#if USE_TDM
  __shared__ __align__(16) bf16 KV[2][32 * DH];
#else
  __shared__ __align__(16) bf16 KV[1][32 * DH];
#endif
  __shared__ __align__(16) bf16 P[8][16 * 32];   // per-wave prob scratch

  v16bf qf[2];
  {
    int r  = q0 + c16;
    int kh = hl << 3;
    const bf16* qp = base + (size_t)r * D;
    qf[0] = cat16(*(const v8bf*)(qp + kh),      *(const v8bf*)(qp + kh + 16));
    qf[1] = cat16(*(const v8bf*)(qp + 32 + kh), *(const v8bf*)(qp + 32 + kh + 16));
  }

  v8f o[4];
#pragma unroll
  for (int j = 0; j < 4; j++) o[j] = {};
  float mrun[8], lrun[8];
#pragma unroll
  for (int e = 0; e < 8; e++){ mrun[e] = -1e30f; lrun[e] = 0.f; }

  int buf = 0;
#if USE_TDM
  if (wave == 0){
    tdm_load_2d(lds_off(KV[0]), base, D, S, DH, 32, D);
    __builtin_amdgcn_s_wait_tensorcnt(0);
  }
  __syncthreads();
#else
  const int cr = tid >> 3;
  const int cc = (tid & 7) << 3;
#endif

  for (int kb = 0; kb < S; kb += 32){
#if USE_TDM
    if (wave == 0 && kb + 32 < S)
      tdm_load_2d(lds_off(KV[buf ^ 1]), base + (size_t)(kb + 32) * D, D, S, DH, 32, D);
#else
    *(v8bf*)(KV[0] + cr * DH + cc) = *(const v8bf*)(base + (size_t)(kb + cr) * D + cc);
    __syncthreads();
#endif
    const bf16* kv = KV[buf];

    // scores: Q(16x64) . K^T -> two 16x16 tiles (keys 0-15, 16-31)
    v8f s0 = {}, s1 = {};
    s0 = wmma_bf16(qf[0], frag_b(kv,               DH, lane), s0);
    s0 = wmma_bf16(qf[1], frag_b(kv + 32,          DH, lane), s0);
    s1 = wmma_bf16(qf[0], frag_b(kv + 16 * DH,      DH, lane), s1);
    s1 = wmma_bf16(qf[1], frag_b(kv + 16 * DH + 32, DH, lane), s1);

    float m0 = mb[kb + c16], m1 = mb[kb + 16 + c16];
    float p0[8], p1[8], mx[8];
#pragma unroll
    for (int e = 0; e < 8; e++){
      p0[e] = s0[e] * 0.125f + m0;     // 1/sqrt(64)
      p1[e] = s1[e] * 0.125f + m1;
      mx[e] = fmaxf(p0[e], p1[e]);
    }
    // row max across the 16 lanes of each half (halves are independent rows)
#pragma unroll
    for (int off = 8; off >= 1; off >>= 1)
#pragma unroll
      for (int e = 0; e < 8; e++) mx[e] = fmaxf(mx[e], __shfl_xor(mx[e], off, 32));

    float rs[8];
#pragma unroll
    for (int e = 0; e < 8; e++){
      float mn   = fmaxf(mrun[e], mx[e]);
      float corr = __expf(mrun[e] - mn);
      mrun[e] = mn;
      p0[e] = __expf(p0[e] - mn);
      p1[e] = __expf(p1[e] - mn);
      rs[e] = p0[e] + p1[e];
      lrun[e] *= corr;
#pragma unroll
      for (int j = 0; j < 4; j++) o[j][e] *= corr;
    }
#pragma unroll
    for (int off = 8; off >= 1; off >>= 1)
#pragma unroll
      for (int e = 0; e < 8; e++) rs[e] += __shfl_xor(rs[e], off, 32);
#pragma unroll
    for (int e = 0; e < 8; e++) lrun[e] += rs[e];

    // re-layout P (C-layout) -> A fragment via per-wave LDS scratch
    bf16* pp = &P[wave][0];
#pragma unroll
    for (int e = 0; e < 8; e++){
      int row = e + 8 * hl;
      pp[row * 32 + c16]      = (bf16)p0[e];
      pp[row * 32 + 16 + c16] = (bf16)p1[e];
    }
    asm volatile("s_wait_dscnt 0" ::: "memory");
    v16bf pf = frag_a(pp, 32, lane);
#pragma unroll
    for (int j = 0; j < 4; j++)
      o[j] = wmma_bf16(pf, frag_b_col(kv + j * 16, DH, lane), o[j]);

#if USE_TDM
    if (wave == 0) __builtin_amdgcn_s_wait_tensorcnt(0);
    __syncthreads();
    buf ^= 1;
#else
    __syncthreads();
#endif
  }

  bf16* ob = attn + (size_t)b * S * D + h * DH;
#pragma unroll
  for (int j = 0; j < 4; j++)
#pragma unroll
    for (int e = 0; e < 8; e++){
      int q = q0 + e + 8 * hl;
      ob[(size_t)q * D + j * 16 + c16] = (bf16)(o[j][e] / lrun[e]);
    }
}

// ------------------------------------------------------ add + layernorm ----
template<bool WITH_BF16>
__global__ __launch_bounds__(256) void add_ln_kernel(
    const float* __restrict__ x, const float* __restrict__ res,
    const float* __restrict__ g, const float* __restrict__ be,
    float* __restrict__ of, bf16* __restrict__ ob, int D)
{
  const int row = blockIdx.x;
  const int tid = threadIdx.x, lane = tid & 31, wave = tid >> 5;
  const float* xr = x + (size_t)row * D;
  const float* rr = res + (size_t)row * D;
  float s = 0.f, s2 = 0.f;
  for (int i = tid; i < D; i += 256){ float v = xr[i] + rr[i]; s += v; s2 += v * v; }
#pragma unroll
  for (int off = 16; off >= 1; off >>= 1){
    s  += __shfl_xor(s,  off, 32);
    s2 += __shfl_xor(s2, off, 32);
  }
  __shared__ float sm[8], sm2[8];
  __shared__ float mu, rstd;
  if (lane == 0){ sm[wave] = s; sm2[wave] = s2; }
  __syncthreads();
  if (tid == 0){
    float ts = 0.f, ts2 = 0.f;
    for (int w = 0; w < 8; w++){ ts += sm[w]; ts2 += sm2[w]; }
    float m = ts / (float)D;
    mu = m;
    rstd = rsqrtf(ts2 / (float)D - m * m + 1e-5f);
  }
  __syncthreads();
  float m = mu, r = rstd;
  for (int i = tid; i < D; i += 256){
    float v = (xr[i] + rr[i] - m) * r * g[i] + be[i];
    of[(size_t)row * D + i] = v;
    if (WITH_BF16) ob[(size_t)row * D + i] = (bf16)v;
  }
}

// ------------------------------------------------------------------ driver --
extern "C" void kernel_launch(void* const* d_in, const int* in_sizes, int n_in,
                              void* d_out, int out_size, void* d_ws, size_t ws_size,
                              hipStream_t stream)
{
  (void)in_sizes; (void)n_in; (void)out_size; (void)ws_size;
  constexpr int Bc = 2, S = 2048, D = 1024, H = 16, DFF = 4096;
  constexpr int M = Bc * S;  // 4096

  const float* hidden = (const float*)d_in[0];
  const float* mask   = (const float*)d_in[1];
  const float* q_w    = (const float*)d_in[2];
  const float* q_b    = (const float*)d_in[3];
  const float* so_w   = (const float*)d_in[4];
  const float* so_b   = (const float*)d_in[5];
  const float* ln1_g  = (const float*)d_in[6];
  const float* ln1_b  = (const float*)d_in[7];
  const float* in_w   = (const float*)d_in[8];
  const float* in_b   = (const float*)d_in[9];
  const float* out_w  = (const float*)d_in[10];
  const float* out_b  = (const float*)d_in[11];
  const float* ln2_g  = (const float*)d_in[12];
  const float* ln2_b  = (const float*)d_in[13];

  bf16* p = (bf16*)d_ws;
  bf16* hsb  = p; p += (size_t)M * D;
  bf16* qwb  = p; p += (size_t)D * D;
  bf16* sowb = p; p += (size_t)D * D;
  bf16* iwb  = p; p += (size_t)DFF * D;
  bf16* owb  = p; p += (size_t)D * DFF;
  bf16* qkvb = p; p += (size_t)M * D;
  bf16* attb = p; p += (size_t)M * D;
  bf16* sob  = p; p += (size_t)M * D;
  bf16* intb = p; p += (size_t)M * DFF;
  float* tmpf = (float*)p;
  float* sof  = tmpf + (size_t)M * D;

  auto cvt = [&](const float* src, bf16* dst, size_t n){
    int n4 = (int)(n / 4);
    cvt_kernel<<<dim3((n4 + 255) / 256), 256, 0, stream>>>(src, dst, n4);
  };
  cvt(hidden, hsb, (size_t)M * D);
  cvt(q_w,   qwb,  (size_t)D * D);
  cvt(so_w,  sowb, (size_t)D * D);
  cvt(in_w,  iwb,  (size_t)DFF * D);
  cvt(out_w, owb,  (size_t)D * DFF);

  // qkv = hs @ q_w^T + q_b  (bf16 out)
  gemm_kernel<0><<<dim3((M / 128) * (D / 128)), 256, 0, stream>>>(hsb, qwb, q_b, qkvb, M, D, D);
  // attention (q=k=v=qkv)
  attention_kernel<<<dim3(Bc * H * (S / 128)), 256, 0, stream>>>(qkvb, mask, attb);
  // self-output dense (f32 out), then +residual & LN1 (keep f32 and bf16)
  gemm_kernel<1><<<dim3((M / 128) * (D / 128)), 256, 0, stream>>>(attb, sowb, so_b, tmpf, M, D, D);
  add_ln_kernel<true><<<dim3(M), 256, 0, stream>>>(tmpf, hidden, ln1_g, ln1_b, sof, sob, D);
  // intermediate dense + exact GELU (bf16 out)
  gemm_kernel<2><<<dim3((M / 128) * (DFF / 128)), 256, 0, stream>>>(sob, iwb, in_b, intb, M, DFF, D);
  // output dense (f32), then +residual & LN2 -> d_out
  gemm_kernel<1><<<dim3((M / 128) * (D / 128)), 256, 0, stream>>>(intb, owb, out_b, tmpf, M, D, DFF);
  add_ln_kernel<false><<<dim3(M), 256, 0, stream>>>(tmpf, sof, ln2_g, ln2_b, (float*)d_out, nullptr, D);
}
